// EncoderProcessDecode_37632503447814
// MI455X (gfx1250) — matrix operand order repaired and checked by reading, not compiled, not run on hardware
//
#include <hip/hip_runtime.h>
#include <hip/hip_bf16.h>

// ---------------------------------------------------------------------------
// EncoderProcessDecode GNN for gfx1250 (MI455X).
// All 128-wide GEMM layers run on v_wmma_f32_16x16x32_bf16 (wave32 WMMA).
// 32-row blocks: each wave computes a 2x2 grid of 16x16 tiles, so every
// K-chunk costs 2 A-frag + 2 B-frag loads for 4 WMMAs.
// ---------------------------------------------------------------------------

#define N_NODES 10000
#define N_EDGES 160000
#define STEPS   6
#define LAT     128
#define EPS_LN  1e-5f

typedef __attribute__((ext_vector_type(16))) __bf16 v16bf;
typedef __attribute__((ext_vector_type(8)))  __bf16 v8bf;
typedef __attribute__((ext_vector_type(8)))  float  v8f;

// ---- bf16 <-> f32 helpers (round-to-nearest-even) -------------------------
__device__ __forceinline__ unsigned short f2bf_u(float f) {
  unsigned u = __builtin_bit_cast(unsigned, f);
  unsigned r = u + 0x7FFFu + ((u >> 16) & 1u);
  return (unsigned short)(r >> 16);
}
__device__ __forceinline__ __bf16 f2bf(float f) {
  unsigned short h = f2bf_u(f);
  return __builtin_bit_cast(__bf16, h);
}
__device__ __forceinline__ float bf2f(__bf16 b) {
  unsigned u = ((unsigned)__builtin_bit_cast(unsigned short, b)) << 16;
  return __builtin_bit_cast(float, u);
}

// Load 16 bf16 as two 16B vectors: lo at p[0..7], hi at p[hioff..hioff+7].
__device__ __forceinline__ v16bf load16(const __bf16* p, int hioff) {
  v8bf lo = *(const v8bf*)(p);
  v8bf hi = *(const v8bf*)(p + hioff);
  v16bf r;
#pragma unroll
  for (int i = 0; i < 8; ++i) { r[i] = lo[i]; r[8 + i] = hi[i]; }
  return r;
}

// ---------------------------------------------------------------------------
// Weight swizzle: fp32 row-major W[K x 128] -> bf16 B-fragment order:
//   [nt(8)][kc(K/32)][lane(32)][elem(16)]
// lane 0-15  : N = nt*16 + lane,    K = kc*32 +     (elem&7) + (elem>>3)*16
// lane 16-31 : N = nt*16 + lane-16, K = kc*32 + 8 + (elem&7) + (elem>>3)*16
// Rows with K >= din (zero padding) are written as 0.
// ---------------------------------------------------------------------------
__global__ void swizzle_w_kernel(const float* __restrict__ src,
                                 unsigned short* __restrict__ dst,
                                 int din, int kchunks) {
  int idx = blockIdx.x * blockDim.x + threadIdx.x;
  int total = kchunks << 12;               // 8*kchunks*32*16
  if (idx >= total) return;
  int e    = idx & 15;
  int lane = (idx >> 4) & 31;
  int rest = idx >> 9;
  int kc   = rest % kchunks;
  int nt   = rest / kchunks;
  int k = kc * 32 + ((e >> 3) << 4) + ((lane >> 4) << 3) + (e & 7);
  int n = nt * 16 + (lane & 15);
  float v = (k < din) ? src[(size_t)k * 128 + n] : 0.f;
  dst[idx] = f2bf_u(v);
}

__global__ void zero_f32_kernel(float* __restrict__ p, int n) {
  int i = blockIdx.x * blockDim.x + threadIdx.x;
  if (i < n) p[i] = 0.f;
}

// ---------------------------------------------------------------------------
// Epilogue helper: write one 16x16 D-tile (8 f32 per lane) with bias (+ReLU).
// C/D layout: lane l, reg r -> M = r + (l>=16 ? 8:0) (+mext), N = ntile*16+(l&15)
// ---------------------------------------------------------------------------
template<bool RELU, bool TOBF>
__device__ __forceinline__ void store_tile(const v8f& acc, int mbase, int n,
                                           float bv, __bf16* __restrict__ hb,
                                           float* __restrict__ hf) {
#pragma unroll
  for (int r = 0; r < 8; ++r) {
    float h = acc[r] + bv;
    if (RELU) h = h > 0.f ? h : 0.f;
    if (TOBF) hb[(mbase + r) * 128 + n] = f2bf(h);
    else      hf[(mbase + r) * 128 + n] = h;
  }
}

// ---------------------------------------------------------------------------
// One dense layer on a 32-row tile: out[32 x 128] = act(xin[32 x KP] @ W + b)
// Block = 128 threads = 4 waves; wave w owns N-tiles {2w, 2w+1} x M-tiles {0,1}.
// ---------------------------------------------------------------------------
template<int KP, bool RELU, bool TOBF>
__device__ __forceinline__ void layer128_m32(const __bf16* __restrict__ xin,
                                             const __bf16* __restrict__ wsw,
                                             const float* __restrict__ bias,
                                             __bf16* __restrict__ hb,
                                             float* __restrict__ hf) {
  const int tid   = threadIdx.x;
  const int wave  = tid >> 5;
  const int lane  = tid & 31;
  const int half8 = (lane >> 4) << 3;   // 0 or 8: A k-offset base AND C row base
  const int nlo   = lane & 15;
  constexpr int KC = KP / 32;
  const int nt0 = wave * 2;
  v8f acc00 = {}, acc01 = {}, acc10 = {}, acc11 = {};
#pragma unroll
  for (int kc = 0; kc < KC; ++kc) {
    // A frags: m = lane&15 (+16 for tile 1); elems 0..7 -> K=kb..kb+7, 8..15 -> K=kb+16..+23
    const __bf16* pa = xin + nlo * KP + kc * 32 + half8;
    v16bf a0 = load16(pa, 16);
    v16bf a1 = load16(pa + 16 * KP, 16);
    // B frags: lane's 16 elements contiguous in swizzled buffer
    const __bf16* pb = wsw + ((((size_t)nt0 * KC + kc) * 32 + lane) << 4);
    v16bf b0 = load16(pb, 8);
    v16bf b1 = load16(pb + ((size_t)(KC * 32) << 4), 8);   // nt0+1 block
    acc00 = __builtin_amdgcn_wmma_f32_16x16x32_bf16(false, a0, false, b0, (short)0, acc00, false, false);
    acc01 = __builtin_amdgcn_wmma_f32_16x16x32_bf16(false, a0, false, b1, (short)0, acc01, false, false);
    acc10 = __builtin_amdgcn_wmma_f32_16x16x32_bf16(false, a1, false, b0, (short)0, acc10, false, false);
    acc11 = __builtin_amdgcn_wmma_f32_16x16x32_bf16(false, a1, false, b1, (short)0, acc11, false, false);
  }
  const int n0 = nt0 * 16 + nlo;
  const int n1 = n0 + 16;
  const float bv0 = bias[n0], bv1 = bias[n1];
  store_tile<RELU, TOBF>(acc00, half8,      n0, bv0, hb, hf);
  store_tile<RELU, TOBF>(acc01, half8,      n1, bv1, hb, hf);
  store_tile<RELU, TOBF>(acc10, half8 + 16, n0, bv0, hb, hf);
  store_tile<RELU, TOBF>(acc11, half8 + 16, n1, bv1, hb, hf);
}

// 3-layer MLP + LayerNorm over last dim (128) for a 32-row tile.
template<int DINP>
__device__ __forceinline__ void mlp_ln32(const __bf16* __restrict__ xin,
                                         __bf16* __restrict__ h1,
                                         __bf16* __restrict__ h2,
                                         float* __restrict__ outf,
                                         float* __restrict__ sred,
                                         const __bf16* w1, const __bf16* w2, const __bf16* w3,
                                         const float* b1, const float* b2, const float* b3,
                                         const float* gamma, const float* beta) {
  const int tid = threadIdx.x;
  layer128_m32<DINP, true,  true >(xin, w1, b1, h1, nullptr);
  __syncthreads();
  layer128_m32<128,  true,  true >(h1,  w2, b2, h2, nullptr);
  __syncthreads();
  layer128_m32<128,  false, false>(h2,  w3, b3, nullptr, outf);
  __syncthreads();
  if (tid < 32) {
    float mu = 0.f;
#pragma unroll 8
    for (int c = 0; c < 128; ++c) mu += outf[tid * 128 + c];
    mu *= (1.f / 128.f);
    float var = 0.f;
#pragma unroll 8
    for (int c = 0; c < 128; ++c) { float d = outf[tid * 128 + c] - mu; var += d * d; }
    var *= (1.f / 128.f);
    sred[tid]      = mu;
    sred[32 + tid] = rsqrtf(var + EPS_LN);
  }
  __syncthreads();
  {
    const int row = tid >> 2;              // 0..31
    const int cb  = (tid & 3) << 5;        // 0,32,64,96
    const float mu = sred[row], rs = sred[32 + row];
#pragma unroll
    for (int c = cb; c < cb + 32; ++c)
      outf[row * 128 + c] = (outf[row * 128 + c] - mu) * rs * gamma[c] + beta[c];
  }
  __syncthreads();
}

// ---------------------------------------------------------------------------
// Encoder kernels (K padded to 32); nrows guards the ragged last node block.
// ---------------------------------------------------------------------------
template<int DIN>
__global__ void __launch_bounds__(128)
encode_kernel(const float* __restrict__ feat, float* __restrict__ out, int nitems,
              const unsigned short* w1u, const unsigned short* w2u, const unsigned short* w3u,
              const float* b1, const float* b2, const float* b3,
              const float* gamma, const float* beta) {
  __shared__ alignas(16) __bf16 xin[32 * 32];
  __shared__ alignas(16) __bf16 h1[32 * 128];
  __shared__ alignas(16) __bf16 h2[32 * 128];
  __shared__ alignas(16) float  outf[32 * 128];
  __shared__ float sred[64];
  const int tid   = threadIdx.x;
  const int r0    = blockIdx.x * 32;
  const int nrows = min(32, nitems - r0);
  for (int i = tid; i < 32 * 32; i += 128) {
    int row = i >> 5, c = i & 31;
    float val = (row < nrows && c < DIN) ? feat[(size_t)(r0 + row) * DIN + c] : 0.f;
    xin[i] = f2bf(val);
  }
  __syncthreads();
  mlp_ln32<32>(xin, h1, h2, outf, sred,
               (const __bf16*)w1u, (const __bf16*)w2u, (const __bf16*)w3u,
               b1, b2, b3, gamma, beta);
  for (int i = tid; i < 32 * 128; i += 128) {
    int row = i >> 7;
    if (row < nrows) out[(size_t)r0 * 128 + i] = outf[i];
  }
}

// ---------------------------------------------------------------------------
// Process: edge MLP (K=384) + residual e += new_e + scatter-add into agg
// N_EDGES % 32 == 0, no guards needed.
// ---------------------------------------------------------------------------
__global__ void __launch_bounds__(128)
proc_edge_kernel(const float* __restrict__ v, float* __restrict__ e,
                 float* __restrict__ agg,
                 const int* __restrict__ senders, const int* __restrict__ receivers,
                 const unsigned short* w1u, const unsigned short* w2u, const unsigned short* w3u,
                 const float* b1, const float* b2, const float* b3,
                 const float* gamma, const float* beta) {
  __shared__ alignas(16) __bf16 xin[32 * 384];
  __shared__ alignas(16) __bf16 h1[32 * 128];
  __shared__ alignas(16) __bf16 h2[32 * 128];
  __shared__ alignas(16) float  outf[32 * 128];
  __shared__ float sred[64];
  __shared__ int ss[32], rr[32];
  const int tid = threadIdx.x;
  const int e0  = blockIdx.x * 32;
  if (tid < 32) { ss[tid] = senders[e0 + tid]; rr[tid] = receivers[e0 + tid]; }
  __syncthreads();
  for (int i = tid; i < 32 * 384; i += 128) {
    int row = i / 384, c = i - row * 384;
    float val;
    if (c < 128)      val = v[(size_t)ss[row] * 128 + c];
    else if (c < 256) val = v[(size_t)rr[row] * 128 + (c - 128)];
    else              val = e[(size_t)(e0 + row) * 128 + (c - 256)];
    xin[i] = f2bf(val);
  }
  __syncthreads();
  mlp_ln32<384>(xin, h1, h2, outf, sred,
                (const __bf16*)w1u, (const __bf16*)w2u, (const __bf16*)w3u,
                b1, b2, b3, gamma, beta);
  const int c = tid;  // 0..127 column, coalesced per row
  for (int row = 0; row < 32; ++row) {
    float ne = outf[row * 128 + c];
    e[(size_t)(e0 + row) * 128 + c] += ne;                 // residual
    atomicAdd(&agg[(size_t)rr[row] * 128 + c], ne);        // segment_sum
  }
}

// ---------------------------------------------------------------------------
// Process: node MLP (K=256, input = [v || agg]) + residual v += new_v
// ---------------------------------------------------------------------------
__global__ void __launch_bounds__(128)
proc_node_kernel(float* __restrict__ v, const float* __restrict__ agg,
                 const unsigned short* w1u, const unsigned short* w2u, const unsigned short* w3u,
                 const float* b1, const float* b2, const float* b3,
                 const float* gamma, const float* beta) {
  __shared__ alignas(16) __bf16 xin[32 * 256];
  __shared__ alignas(16) __bf16 h1[32 * 128];
  __shared__ alignas(16) __bf16 h2[32 * 128];
  __shared__ alignas(16) float  outf[32 * 128];
  __shared__ float sred[64];
  const int tid   = threadIdx.x;
  const int n0    = blockIdx.x * 32;
  const int nrows = min(32, N_NODES - n0);
  for (int i = tid; i < 32 * 256; i += 128) {
    int row = i >> 8, c = i & 255;
    float val = 0.f;
    if (row < nrows)
      val = (c < 128) ? v[(size_t)(n0 + row) * 128 + c]
                      : agg[(size_t)(n0 + row) * 128 + (c - 128)];
    xin[i] = f2bf(val);
  }
  __syncthreads();
  mlp_ln32<256>(xin, h1, h2, outf, sred,
                (const __bf16*)w1u, (const __bf16*)w2u, (const __bf16*)w3u,
                b1, b2, b3, gamma, beta);
  const int c = tid;
  for (int row = 0; row < nrows; ++row)
    v[(size_t)(n0 + row) * 128 + c] += outf[row * 128 + c];
}

// ---------------------------------------------------------------------------
// Decoder: two WMMA layers, scalar 128->3 layer, LayerNorm over 3 outputs
// ---------------------------------------------------------------------------
__global__ void __launch_bounds__(128)
decode_kernel(const float* __restrict__ v, float* __restrict__ out,
              const unsigned short* w1u, const unsigned short* w2u,
              const float* b1, const float* b2,
              const float* __restrict__ W3, const float* __restrict__ b3,
              const float* gamma, const float* beta) {
  __shared__ alignas(16) __bf16 xin[32 * 128];
  __shared__ alignas(16) __bf16 h1[32 * 128];
  __shared__ alignas(16) __bf16 h2[32 * 128];
  __shared__ float out3[32 * 3];
  const int tid   = threadIdx.x;
  const int n0    = blockIdx.x * 32;
  const int nrows = min(32, N_NODES - n0);
  for (int i = tid; i < 32 * 128; i += 128) {
    int row = i >> 7;
    xin[i] = f2bf(row < nrows ? v[(size_t)n0 * 128 + i] : 0.f);
  }
  __syncthreads();
  layer128_m32<128, true, true>(xin, (const __bf16*)w1u, b1, h1, nullptr);
  __syncthreads();
  layer128_m32<128, true, true>(h1, (const __bf16*)w2u, b2, h2, nullptr);
  __syncthreads();
  if (tid < 96) {
    int row = tid / 3, oc = tid - row * 3;
    float acc = b3[oc];
    for (int k = 0; k < 128; ++k) acc += bf2f(h2[row * 128 + k]) * W3[k * 3 + oc];
    out3[row * 3 + oc] = acc;
  }
  __syncthreads();
  if (tid < nrows) {
    float a = out3[tid * 3 + 0], b = out3[tid * 3 + 1], c = out3[tid * 3 + 2];
    float mu = (a + b + c) * (1.f / 3.f);
    float va = ((a - mu) * (a - mu) + (b - mu) * (b - mu) + (c - mu) * (c - mu)) * (1.f / 3.f);
    float rs = rsqrtf(va + EPS_LN);
    out[(size_t)(n0 + tid) * 3 + 0] = (a - mu) * rs * gamma[0] + beta[0];
    out[(size_t)(n0 + tid) * 3 + 1] = (b - mu) * rs * gamma[1] + beta[1];
    out[(size_t)(n0 + tid) * 3 + 2] = (c - mu) * rs * gamma[2] + beta[2];
  }
}

// ---------------------------------------------------------------------------
// Host-side orchestration
// ---------------------------------------------------------------------------
extern "C" void kernel_launch(void* const* d_in, const int* in_sizes, int n_in,
                              void* d_out, int out_size, void* d_ws, size_t ws_size,
                              hipStream_t stream) {
  (void)in_sizes; (void)n_in; (void)out_size; (void)ws_size;
  const float* node_feat = (const float*)d_in[0];
  const float* edge_feat = (const float*)d_in[1];
  const int*   senders   = (const int*)d_in[2];
  const int*   receivers = (const int*)d_in[3];
  // param tuples: (W1,b1,W2,b2,W3,b3,gamma,beta)
  // enc_node: 4..11  enc_edge: 12..19  proc_edge: 20..27  proc_node: 28..35  dec: 36..43
  auto F = [&](int i) { return (const float*)d_in[i]; };

  // ---- workspace carve-out ----
  char* ws = (char*)d_ws;
  float* v   = (float*)ws;                      ws += (size_t)N_NODES * LAT * 4;
  float* e   = (float*)ws;                      ws += (size_t)N_EDGES * LAT * 4;
  float* agg = (float*)ws;                      ws += (size_t)N_NODES * LAT * 4;
  unsigned short* wb = (unsigned short*)ws;     // bf16 swizzled weights (~2 MB)

  size_t woff = 0;
  auto swz = [&](const float* src, int din, int kp) -> const unsigned short* {
    unsigned short* dst = wb + woff;
    int total = kp * 128;
    swizzle_w_kernel<<<(total + 255) / 256, 256, 0, stream>>>(src, dst, din, kp / 32);
    woff += (size_t)total;
    return dst;
  };

  const unsigned short* en_w1 = swz(F(4), 12, 32);
  const unsigned short* en_w2 = swz(F(6), 128, 128);
  const unsigned short* en_w3 = swz(F(8), 128, 128);
  const unsigned short* ee_w1 = swz(F(12), 7, 32);
  const unsigned short* ee_w2 = swz(F(14), 128, 128);
  const unsigned short* ee_w3 = swz(F(16), 128, 128);
  const unsigned short *pe_w1[STEPS], *pe_w2[STEPS], *pe_w3[STEPS];
  const unsigned short *pn_w1[STEPS], *pn_w2[STEPS], *pn_w3[STEPS];
  for (int s = 0; s < STEPS; ++s) {
    pe_w1[s] = swz(F(20) + (size_t)s * 384 * 128, 384, 384);
    pe_w2[s] = swz(F(22) + (size_t)s * 128 * 128, 128, 128);
    pe_w3[s] = swz(F(24) + (size_t)s * 128 * 128, 128, 128);
    pn_w1[s] = swz(F(28) + (size_t)s * 256 * 128, 256, 256);
    pn_w2[s] = swz(F(30) + (size_t)s * 128 * 128, 128, 128);
    pn_w3[s] = swz(F(32) + (size_t)s * 128 * 128, 128, 128);
  }
  const unsigned short* de_w1 = swz(F(36), 128, 128);
  const unsigned short* de_w2 = swz(F(38), 128, 128);

  const int NODE_BLOCKS = (N_NODES + 31) / 32;  // 313 (last block ragged)
  const int EDGE_BLOCKS = N_EDGES / 32;         // 5000

  // ---- encoders ----
  encode_kernel<12><<<NODE_BLOCKS, 128, 0, stream>>>(
      node_feat, v, N_NODES, en_w1, en_w2, en_w3, F(5), F(7), F(9), F(10), F(11));
  encode_kernel<7><<<EDGE_BLOCKS, 128, 0, stream>>>(
      edge_feat, e, N_EDGES, ee_w1, ee_w2, ee_w3, F(13), F(15), F(17), F(18), F(19));

  // ---- message-passing steps ----
  const int AGG_N = N_NODES * LAT;
  for (int s = 0; s < STEPS; ++s) {
    zero_f32_kernel<<<(AGG_N + 255) / 256, 256, 0, stream>>>(agg, AGG_N);
    proc_edge_kernel<<<EDGE_BLOCKS, 128, 0, stream>>>(
        v, e, agg, senders, receivers,
        pe_w1[s], pe_w2[s], pe_w3[s],
        F(21) + s * 128, F(23) + s * 128, F(25) + s * 128,
        F(26) + s * 128, F(27) + s * 128);
    proc_node_kernel<<<NODE_BLOCKS, 128, 0, stream>>>(
        v, agg,
        pn_w1[s], pn_w2[s], pn_w3[s],
        F(29) + s * 128, F(31) + s * 128, F(33) + s * 128,
        F(34) + s * 128, F(35) + s * 128);
  }

  // ---- decoder ----
  decode_kernel<<<NODE_BLOCKS, 128, 0, stream>>>(
      v, (float*)d_out, de_w1, de_w2, F(37), F(39), F(40), F(41), F(42), F(43));
}
